// WeightedFusionLayer_29712583753815
// MI455X (gfx1250) — matrix operand order repaired and checked by reading, not compiled, not run on hardware
//
#include <hip/hip_runtime.h>

#define BATCH 65536
#define SEQ 49
#define CH 19
#define BB 16                            // batches per workgroup
#define ROWS_PER_WG (BB * SEQ)           // 784
#define TILES_PER_WG (ROWS_PER_WG / 16)  // 49
#define WAVES 8
#define THREADS 256

#define LOG2E 1.44269504088896340736f

typedef __attribute__((ext_vector_type(16))) __bf16 v16bf;
typedef __attribute__((ext_vector_type(8)))  float  v8f;

union BFV { v16bf v; unsigned int u[8]; };

// ---- accurate (RNE) pack: used once for the weight operands ----
__device__ __forceinline__ unsigned short f2bf(float f) {
    unsigned int u = __float_as_uint(f);
    unsigned int r = u + 0x7FFFu + ((u >> 16) & 1u);
    return (unsigned short)(r >> 16);
}
__device__ __forceinline__ unsigned int pk_rne(float lo, float hi) {
    return ((unsigned int)f2bf(hi) << 16) | (unsigned int)f2bf(lo);
}

// ---- fast pack for the streaming data: single v_perm_b32 (truncate) ----
#if __has_builtin(__builtin_amdgcn_perm)
__device__ __forceinline__ unsigned int pk_fast(float lo, float hi) {
    // D.b0=lo.b2 D.b1=lo.b3 D.b2=hi.b2 D.b3=hi.b3  ({S0=hi,S1=lo} byte select)
    return __builtin_amdgcn_perm(__float_as_uint(hi), __float_as_uint(lo), 0x07060302u);
}
#else
__device__ __forceinline__ unsigned int pk_fast(float lo, float hi) {
    return pk_rne(lo, hi);
}
#endif

#if __has_builtin(__builtin_amdgcn_exp2f)
__device__ __forceinline__ float fast_exp2(float x) { return __builtin_amdgcn_exp2f(x); }
#else
__device__ __forceinline__ float fast_exp2(float x) { return exp2f(x); }
#endif

// exchange lane r <-> lane r+16 (ds_swizzle group-of-32: xor=0x10, and=0x1f)
__device__ __forceinline__ float xor16(float x) {
    return __int_as_float(__builtin_amdgcn_ds_swizzle(__float_as_int(x), 0x401F));
}

__global__ __launch_bounds__(THREADS)
void fused_gate_kernel(const float* __restrict__ Hf, const float* __restrict__ Hh,
                       const float* __restrict__ Wa, const float* __restrict__ Wb,
                       const float* __restrict__ Wout, const float* __restrict__ bias,
                       float* __restrict__ out)
{
    // W^T (pre-scaled by log2e) packed in A-operand layout: [matrix][mtile][lane][8 dwords]
    __shared__ __align__(32) unsigned int sWOp[2][2][32][8];   // 4 KB
    __shared__ float sWout[SEQ * CH];                          // 3724 B
    __shared__ float sAcc[BB];

    const int tid = threadIdx.x;
    const int wid = tid >> 5;
    const int ln  = tid & 31;
    const int r   = ln & 15;      // this lane's row within the tile (D-matrix N)
    const int h   = ln >> 4;      // lane half
    const int bb0 = blockIdx.x * BB;

    // ---------------- one-time LDS setup ----------------
    if (tid < BB) sAcc[tid] = 0.0f;
    for (int i = tid; i < SEQ * CH; i += THREADS) sWout[i] = Wout[i];

    #pragma unroll
    for (int q = 0; q < 4; ++q) {
        int idx = tid * 4 + q;            // 1024 packed dwords total
        int mat = idx >> 9;
        int rem = idx & 511;
        int mt  = rem >> 8;               // M-tile (channel block)
        int r2  = rem & 255;
        int l   = r2 >> 3;                // lane
        int dd  = r2 & 7;                 // dword within lane
        // A operand, contiguous k order used here: lanes0-15 K=0..15 (dw0-7); lanes16-31 K=16..31
        int kb  = (l >> 4) * 16 + 2 * dd;
        int d   = mt * 16 + (l & 15);     // A-matrix row M = output channel
        const float* W = mat ? Wb : Wa;
        // A[M=d][K=c] = W^T[d][c] * log2e  -> logits directly in exp2 domain
        float w0 = (kb     < CH && d < CH) ? W[kb * CH + d]       * LOG2E : 0.0f;
        float w1 = (kb + 1 < CH && d < CH) ? W[(kb + 1) * CH + d] * LOG2E : 0.0f;
        sWOp[mat][mt][l][dd] = pk_rne(w0, w1);
    }
    __syncthreads();

    // wave-invariant A operands (two ds_load_b128 each)
    v16bf waT0 = *(const v16bf*)&sWOp[0][0][ln][0];
    v16bf waT1 = *(const v16bf*)&sWOp[0][1][ln][0];
    v16bf wbT0 = *(const v16bf*)&sWOp[1][0][ln][0];
    v16bf wbT1 = *(const v16bf*)&sWOp[1][1][ln][0];

    const float NEG_INF = -__builtin_inff();

    // incremental row bookkeeping: rloc advances by 128 each iteration
    int rloc = wid * 16 + r;
    int sidx = rloc % SEQ;
    int bloc = rloc / SEQ;
    const float* rf = Hf + ((size_t)bb0 * SEQ + rloc) * CH;
    const float* rh = Hh + ((size_t)bb0 * SEQ + rloc) * CH;

    for (int t = wid; t < TILES_PER_WG; t += WAVES) {
        // ---- load this row's channels for the B operand (K = c) ----
        // half 0: c = 0..15 ; half 1: c = 16..18 then zero pad
        float ef[16], eh[16];
        if (h == 0) {
            #pragma unroll
            for (int i = 0; i < 16; ++i) { ef[i] = rf[i]; eh[i] = rh[i]; }
        } else {
            ef[0] = rf[16]; ef[1] = rf[17]; ef[2] = rf[18];
            eh[0] = rh[16]; eh[1] = rh[17]; eh[2] = rh[18];
            #pragma unroll
            for (int i = 3; i < 16; ++i) { ef[i] = 0.0f; eh[i] = 0.0f; }
        }

        BFV bF, bH;
        #pragma unroll
        for (int dd = 0; dd < 8; ++dd) {
            bF.u[dd] = pk_fast(ef[2*dd], ef[2*dd + 1]);
            bH.u[dd] = pk_fast(eh[2*dd], eh[2*dd + 1]);
        }

        // D = (log2e*W^T) x H^T : lane holds row r, channels M = mtile*16 + h*8 + v
        v8f z = {};
        v8f la0 = __builtin_amdgcn_wmma_f32_16x16x32_bf16(false, waT0, false, bF.v, (short)0, z, false, false);
        v8f la1 = __builtin_amdgcn_wmma_f32_16x16x32_bf16(false, waT1, false, bF.v, (short)0, z, false, false);
        v8f lb0 = __builtin_amdgcn_wmma_f32_16x16x32_bf16(false, wbT0, false, bH.v, (short)0, z, false, false);
        v8f lb1 = __builtin_amdgcn_wmma_f32_16x16x32_bf16(false, wbT1, false, bH.v, (short)0, z, false, false);

        // ---- cross-half exchange of raw H values for the fusion step ----
        // half0 sends c=8..15, receives c=16..18 ; half1 sends c=16..18, receives c=8..15
        float exf[8], exh[8];
        #pragma unroll
        for (int i = 0; i < 8; ++i) {
            // half1's ef[i] is already 0 for i>=3, so no extra masking needed
            exf[i] = xor16((h == 0) ? ef[8 + i] : ef[i]);
            exh[i] = xor16((h == 0) ? eh[8 + i] : eh[i]);
        }

        // ---- channel softmax in exp2 domain, lane-local + one SWAPX16 combine ----
        float ma = la0[0], mb = lb0[0];
        #pragma unroll
        for (int v = 1; v < 8; ++v) { ma = fmaxf(ma, la0[v]); mb = fmaxf(mb, lb0[v]); }
        #pragma unroll
        for (int v = 0; v < 3; ++v) {
            ma = fmaxf(ma, (h == 0) ? la1[v] : NEG_INF);
            mb = fmaxf(mb, (h == 0) ? lb1[v] : NEG_INF);
        }
        ma = fmaxf(ma, xor16(ma));
        mb = fmaxf(mb, xor16(mb));

        float ea0[8], eb0[8], ea1[3], eb1[3];
        float sa = 0.0f, sb = 0.0f;
        #pragma unroll
        for (int v = 0; v < 8; ++v) {
            ea0[v] = fast_exp2(la0[v] - ma); sa += ea0[v];
            eb0[v] = fast_exp2(lb0[v] - mb); sb += eb0[v];
        }
        #pragma unroll
        for (int v = 0; v < 3; ++v) {
            ea1[v] = (h == 0) ? fast_exp2(la1[v] - ma) : 0.0f; sa += ea1[v];
            eb1[v] = (h == 0) ? fast_exp2(lb1[v] - mb) : 0.0f; sb += eb1[v];
        }
        sa += xor16(sa);
        sb += xor16(sb);
        const float ia = __builtin_amdgcn_rcpf(sa);
        const float ib = __builtin_amdgcn_rcpf(sb);

        // ---- gated blend * W_out: un-normalized accumulation, normalize once ----
        const float* wrow = &sWout[sidx * CH];
        float pa = 0.0f, pb = 0.0f;
        #pragma unroll
        for (int v = 0; v < 8; ++v) {
            // channel d = h*8 + v ; H[r][d]: half0 has it in ef, half1 got it via exchange
            float wv  = wrow[h * 8 + v];
            float hfw = ((h == 0) ? ef[v] : exf[v]) * wv;
            float hhw = ((h == 0) ? eh[v] : exh[v]) * wv;
            pa = fmaf(ea0[v], hfw, pa);
            pb = fmaf(eb0[v], hhw, pb);
        }
        #pragma unroll
        for (int v = 0; v < 3; ++v) {
            // channels 16..18 (half0 only; ea1/eb1 are zero on half1)
            float wv = wrow[16 + v];
            pa = fmaf(ea1[v], exf[v] * wv, pa);
            pb = fmaf(eb1[v], exh[v] * wv, pb);
        }
        float part = ia * pa + ib * pb;
        part += xor16(part);
        if (h == 0) atomicAdd(&sAcc[bloc], part);   // ds_add_f32, one per row

        // ---- advance 128 rows: rloc+128 = (bloc+2)*49 + (sidx+30), wrap once ----
        rf += 128 * CH;
        rh += 128 * CH;
        sidx += 30; bloc += 2;
        if (sidx >= SEQ) { sidx -= SEQ; ++bloc; }
    }

    __syncthreads();
    if (tid < BB) {
        float u = sAcc[tid] + bias[0];
        out[bb0 + tid] = 1.0f / (1.0f + fast_exp2(-u * LOG2E));
    }
}

extern "C" void kernel_launch(void* const* d_in, const int* in_sizes, int n_in,
                              void* d_out, int out_size, void* d_ws, size_t ws_size,
                              hipStream_t stream) {
    const float* Hf   = (const float*)d_in[0];
    const float* Hh   = (const float*)d_in[1];
    const float* Wa   = (const float*)d_in[2];
    const float* Wb   = (const float*)d_in[3];
    const float* Wout = (const float*)d_in[4];
    const float* bias = (const float*)d_in[5];
    float* out = (float*)d_out;

    dim3 grid(BATCH / BB);     // 4096 workgroups
    dim3 block(THREADS);       // 8 waves (wave32)
    hipLaunchKernelGGL(fused_gate_kernel, grid, block, 0, stream,
                       Hf, Hh, Wa, Wb, Wout, bias, out);
}
